// BiLSTM_CRF_57380763074842
// MI455X (gfx1250) — compile-verified, matrix operand order checked by reading, not run on hardware
//
#include <hip/hip_runtime.h>
#include <hip/hip_bf16.h>
#include <stdint.h>

#define SEQ    8192
#define EMBED  256
#define H2     256
#define GATES  1024   /* 4*H2 per direction */
#define GCAT   2048   /* both directions    */
#define HID    512
#define NTAG   16
#define NEGV   (-10000.0f)

#define WROW_PAD_BYTES 520          /* 512B bf16 row + 8B pad -> bank-conflict-free */
#define WLDS_ROWS      512          /* rows of W_hh resident in LDS                 */
#define WLDS_BYTES     (WLDS_ROWS * WROW_PAD_BYTES)          /* 266240 B */
#define SCAN_LDS_BYTES (WLDS_BYTES + GATES * 4 + H2 * 4)     /* + z[1024], h[256] */
#define VIT_LDS_BYTES  (SEQ * 2 * 4)                         /* 64KB nibble bptrs */

typedef __attribute__((ext_vector_type(16))) __bf16 v16bf;
typedef __attribute__((ext_vector_type(8)))  float  v8f;
typedef __attribute__((ext_vector_type(4)))  unsigned int u32x4;
typedef __attribute__((ext_vector_type(8)))  int i32x8;
typedef __attribute__((ext_vector_type(4)))  int i32x4;

// Build the 16-half A/B operand from two contiguous 16-byte bf16 runs
// (K = off_lo..+7 and K = off_hi..+7) per the CDNA5 16-bit WMMA layouts.
static __device__ __forceinline__ v16bf load_bf16_pair(const __bf16* row,
                                                       int off_lo, int off_hi) {
  union { unsigned int u[8]; v16bf v; } r;
  const unsigned int* p0 = (const unsigned int*)(row + off_lo);
  const unsigned int* p1 = (const unsigned int*)(row + off_hi);
#pragma unroll
  for (int i = 0; i < 4; ++i) { r.u[i] = p0[i]; r.u[4 + i] = p1[i]; }
  return r.v;
}

static __device__ __forceinline__ float sigf(float x) {
  return 1.0f / (1.0f + __expf(-x));
}
// exact bf16 -> f32 (bf16 is the top half of an f32)
static __device__ __forceinline__ float bflo(unsigned int u) {
  return __uint_as_float(u << 16);
}
static __device__ __forceinline__ float bfhi(unsigned int u) {
  return __uint_as_float(u & 0xffff0000u);
}

// ---------------------------------------------------------------------------
// Kernel 1: weight prep.
//   wcat[2048][256]   = {w_ih_f ; w_ih_b}          (bf16)
//   whh_bf[2048][256] = {w_hh_f ; w_hh_b}          (bf16)  <- halves scan traffic
//   wout[16][512]     = W_out                      (bf16)
//   bcat[2048]        = {b_ih_f+b_hh_f ; b_ih_b+b_hh_b}
// ---------------------------------------------------------------------------
__global__ void prep_kernel(const float* __restrict__ w_ih_f,
                            const float* __restrict__ w_ih_b,
                            const float* __restrict__ w_hh_f,
                            const float* __restrict__ w_hh_b,
                            const float* __restrict__ b_ih_f,
                            const float* __restrict__ b_hh_f,
                            const float* __restrict__ b_ih_b,
                            const float* __restrict__ b_hh_b,
                            const float* __restrict__ w_out,
                            __bf16* __restrict__ wcat,
                            __bf16* __restrict__ whh_bf,
                            float*  __restrict__ bcat,
                            __bf16* __restrict__ wout) {
  int idx = blockIdx.x * blockDim.x + threadIdx.x;
  const int NW  = GCAT * EMBED;          // 524288
  const int NHH = GCAT * EMBED;          // 524288
  const int NO  = NTAG * HID;            // 8192
  if (idx < NW) {
    int row = idx >> 8, k = idx & 255;
    float v = (row < GATES) ? w_ih_f[row * EMBED + k]
                            : w_ih_b[(row - GATES) * EMBED + k];
    wcat[idx] = (__bf16)v;
  } else if (idx < NW + NHH) {
    int i2 = idx - NW;
    int row = i2 >> 8, k = i2 & 255;
    float v = (row < GATES) ? w_hh_f[row * H2 + k]
                            : w_hh_b[(row - GATES) * H2 + k];
    whh_bf[i2] = (__bf16)v;
  } else if (idx < NW + NHH + NO) {
    int i2 = idx - NW - NHH;
    wout[i2] = (__bf16)w_out[i2];
  } else if (idx < NW + NHH + NO + GCAT) {
    int g = idx - NW - NHH - NO;
    bcat[g] = (g < GATES) ? (b_ih_f[g] + b_hh_f[g])
                          : (b_ih_b[g - GATES] + b_hh_b[g - GATES]);
  }
}

// ---------------------------------------------------------------------------
// Kernel 2: embedding gather -> bf16 rows
// ---------------------------------------------------------------------------
__global__ void gather_kernel(const int* __restrict__ sentence,
                              const float* __restrict__ embed,
                              __bf16* __restrict__ emb_bf) {
  int idx = blockIdx.x * blockDim.x + threadIdx.x;   // SEQ*EMBED threads
  int s = idx >> 8, k = idx & 255;
  int tok = sentence[s];
  emb_bf[idx] = (__bf16)embed[tok * EMBED + k];
}

// ---------------------------------------------------------------------------
// Kernel 3: WMMA GEMM  Z_in[s][g] = emb[s]·wcat[g] + bcat[g]
//   [8192x256]x[256x2048]; one 16x16 tile per wave; K-loop 8x wmma bf16.
// ---------------------------------------------------------------------------
__global__ void zin_gemm_kernel(const __bf16* __restrict__ emb,
                                const __bf16* __restrict__ wcat,
                                const float*  __restrict__ bcat,
                                float* __restrict__ zin) {
  int wave = (blockIdx.x * blockDim.x + threadIdx.x) >> 5;  // 0..65535
  int lane = threadIdx.x & 31;
  int s0 = (wave >> 7) * 16;            // 512 row tiles
  int j0 = (wave & 127) * 16;           // 128 col tiles
  int m    = lane & 15;
  int koff = (lane < 16) ? 0 : 8;
  const __bf16* arow = emb  + (s0 + m) * EMBED;
  const __bf16* brow = wcat + (j0 + m) * EMBED;
  v8f c = {};
#pragma unroll
  for (int kc = 0; kc < EMBED; kc += 32) {
    v16bf a = load_bf16_pair(arow, kc + koff, kc + 16 + koff);
    v16bf b = load_bf16_pair(brow, kc + koff, kc + 16 + koff);
    c = __builtin_amdgcn_wmma_f32_16x16x32_bf16(false, a, false, b,
                                                (short)0, c, false, false);
  }
  int col   = j0 + (lane & 15);
  int rbase = s0 + ((lane < 16) ? 0 : 8);
  float bias = bcat[col];
#pragma unroll
  for (int r = 0; r < 8; ++r)
    zin[(rbase + r) * GCAT + col] = c[r] + bias;
}

// ---------------------------------------------------------------------------
// Kernel 4: sequential LSTM scan, one workgroup (WGP) per direction.
//   W_hh rows 0..511 (bf16, 256KB) live in LDS with an 8B row pad so that
//   32 consecutive rows map to distinct banks -> conflict-free ds_load_b64.
//   Rows 512..1023 stream from L2 as bf16 (256KB/step, 4x less than round 1).
//   LDS image is first requested via the Tensor Data Mover (pad_enable gives
//   the 520B stride), then rewritten by a plain cooperative copy so the data
//   is correct independent of D# field interpretation.
// ---------------------------------------------------------------------------
__global__ void __launch_bounds__(1024)
lstm_scan_kernel(const float* __restrict__ zin,
                 const __bf16* __restrict__ whh_bf,
                 const float* __restrict__ h0,
                 const float* __restrict__ c0,
                 __bf16* __restrict__ hcat) {
  extern __shared__ char smem[];                    // no static LDS: base off 0
  float* z_lds = (float*)(smem + WLDS_BYTES);       // [GATES]
  float* h_lds = z_lds + GATES;                     // [H2]
  const int dir = blockIdx.x;                       // 0 = fwd, 1 = bwd
  const int t   = threadIdx.x;                      // 0..1023
  const __bf16* whh_dir = whh_bf + (size_t)dir * GATES * EMBED;

  // ---- TDM preload request: 2D tile 256x512 bf16, LDS pad 2 dwords / 512B ----
  if ((threadIdx.x >> 5) == 0) {
    unsigned long long ga = (unsigned long long)(uintptr_t)whh_dir;
    u32x4 g0;
    g0[0] = 1u;                                        // count=1
    g0[1] = 0u;                                        // lds_addr = dynamic base
    g0[2] = (unsigned int)(ga & 0xffffffffu);          // global_addr lo
    g0[3] = (unsigned int)((ga >> 32) & 0x01ffffffu) | (2u << 30);  // hi | type=2
    i32x8 g1;
    g1[0] = (1 << 16) | (1 << 20) | (6 << 22) | (1 << 25);
    //        data=2B   pad_en      128dw ivl    2dw pad
    g1[1] = (int)(256u << 16);   // tensor_dim0 = 256
    g1[2] = (int)(512u << 16);   // tensor_dim1 = 512
    g1[3] = (int)(256u << 16);   // tile_dim0   = 256
    g1[4] = 512;                 // tile_dim1   = 512
    g1[5] = 256;                 // tensor_dim0_stride = 256
    g1[6] = 0; g1[7] = 0;
    i32x4 gz; gz[0] = gz[1] = gz[2] = gz[3] = 0;
#if __clang_major__ >= 23
    i32x8 g5; for (int i = 0; i < 8; ++i) g5[i] = 0;
    __builtin_amdgcn_tensor_load_to_lds(g0, g1, gz, gz, g5, 0);
#else
    __builtin_amdgcn_tensor_load_to_lds(g0, g1, gz, gz, 0);
#endif
    __builtin_amdgcn_s_wait_tensorcnt((short)0);
  }
  __syncthreads();
  // ---- belt-and-braces: rewrite identical padded layout cooperatively ----
  {
    const unsigned int* src = (const unsigned int*)whh_dir;  // 128 uints/row
    for (int i = t; i < WLDS_ROWS * 128; i += 1024) {
      int row = i >> 7, k2 = i & 127;
      *(unsigned int*)(smem + row * WROW_PAD_BYTES + (k2 << 2)) = src[i];
    }
  }

  float c = 0.0f;
  if (t < H2) {
    h_lds[t] = h0[dir * H2 + t];
    c        = c0[dir * H2 + t];
  }
  __syncthreads();

  const char* wrow = smem + (size_t)t * WROW_PAD_BYTES;               // t < 512
  const uint4* wg  = (const uint4*)(whh_dir + (size_t)t * H2);        // t >= 512

  for (int step = 0; step < SEQ; ++step) {
    int s = (dir == 0) ? step : (SEQ - 1 - step);
    float z = zin[(size_t)s * GCAT + dir * GATES + t];  // includes both biases
    if (t < WLDS_ROWS) {
      // gate rows i/f: weights from LDS (conflict-free b64 reads)
#pragma unroll 8
      for (int k = 0; k < 64; ++k) {
        unsigned long long u = *(const unsigned long long*)(wrow + (k << 3));
        unsigned int u0 = (unsigned int)u, u1 = (unsigned int)(u >> 32);
        const float* hp = h_lds + (k << 2);
        z += bflo(u0) * hp[0] + bfhi(u0) * hp[1]
           + bflo(u1) * hp[2] + bfhi(u1) * hp[3];
      }
    } else {
      // gate rows g/o: bf16 stream from L2 (b128 loads)
      __builtin_prefetch(wg, 0, 1);
#pragma unroll 4
      for (int j = 0; j < 32; ++j) {
        uint4 u = wg[j];
        const float* hp = h_lds + (j << 3);
        z += bflo(u.x) * hp[0] + bfhi(u.x) * hp[1]
           + bflo(u.y) * hp[2] + bfhi(u.y) * hp[3]
           + bflo(u.z) * hp[4] + bfhi(u.z) * hp[5]
           + bflo(u.w) * hp[6] + bfhi(u.w) * hp[7];
      }
    }
    z_lds[t] = z;
    __syncthreads();
    if (t < H2) {
      float zi = z_lds[t];
      float zf = z_lds[H2 + t];
      float zg = z_lds[2 * H2 + t];
      float zo = z_lds[3 * H2 + t];
      c = sigf(zf) * c + sigf(zi) * tanhf(zg);
      float h = sigf(zo) * tanhf(c);
      h_lds[t] = h;
      hcat[(size_t)s * HID + dir * H2 + t] = (__bf16)h;
    }
    __syncthreads();
  }
}

// ---------------------------------------------------------------------------
// Kernel 5: WMMA projection  feats = hcat·W_outᵀ + b_out
//   [8192x512]x[512x16]; one 16-row tile per wave, N = one 16-tile.
// ---------------------------------------------------------------------------
__global__ void proj_kernel(const __bf16* __restrict__ hcat,
                            const __bf16* __restrict__ wout,
                            const float*  __restrict__ bout,
                            float* __restrict__ feats) {
  int wave = (blockIdx.x * blockDim.x + threadIdx.x) >> 5;  // 0..511
  int lane = threadIdx.x & 31;
  int s0   = wave * 16;
  int m    = lane & 15;
  int koff = (lane < 16) ? 0 : 8;
  const __bf16* arow = hcat + (s0 + m) * HID;
  const __bf16* brow = wout + m * HID;
  v8f c = {};
#pragma unroll
  for (int kc = 0; kc < HID; kc += 32) {
    v16bf a = load_bf16_pair(arow, kc + koff, kc + 16 + koff);
    v16bf b = load_bf16_pair(brow, kc + koff, kc + 16 + koff);
    c = __builtin_amdgcn_wmma_f32_16x16x32_bf16(false, a, false, b,
                                                (short)0, c, false, false);
  }
  int col   = lane & 15;
  int rbase = s0 + ((lane < 16) ? 0 : 8);
  float bias = bout[col];
#pragma unroll
  for (int r = 0; r < 8; ++r)
    feats[(rbase + r) * NTAG + col] = c[r] + bias;
}

// ---------------------------------------------------------------------------
// Kernel 6: Viterbi scan + backtrace (single wave).
//   Back-pointers are 4-bit; 16 of them are packed into one 64-bit word per
//   step via 3x shfl_xor and kept in 64KB of LDS, so the 8192-step dependent
//   backtrace chases LDS latency instead of L2 latency.
//   d_out[0] = path_score; d_out[1..SEQ] = tags (as float).
// ---------------------------------------------------------------------------
__global__ void viterbi_kernel(const float* __restrict__ feats,
                               const float* __restrict__ transitions,
                               float* __restrict__ out) {
  extern __shared__ unsigned int bpw[];   // [SEQ*2] nibble-packed back-ptrs
  __shared__ float fv[NTAG];
  __shared__ float term[NTAG];
  const int n = threadIdx.x;
  float tr[NTAG];
  if (n < NTAG) {
#pragma unroll
    for (int p = 0; p < NTAG; ++p) tr[p] = transitions[n * NTAG + p];
    fv[n] = (n == 0) ? 0.0f : NEGV;       // START = 0
  }
  __syncthreads();

  for (int s = 0; s < SEQ; ++s) {
    float nf = 0.0f;
    int   arg = 0;
    if (n < NTAG) {
      float best = fv[0] + tr[0];
#pragma unroll
      for (int p = 1; p < NTAG; ++p) {
        float v = fv[p] + tr[p];
        if (v > best) { best = v; arg = p; }
      }
      nf = best + feats[s * NTAG + n];
    }
    // pack 8 nibbles per 32-bit word within each half of the 16 tag lanes
    int v = (n < NTAG) ? (arg << ((n & 7) * 4)) : 0;
    v |= __shfl_xor(v, 1);
    v |= __shfl_xor(v, 2);
    v |= __shfl_xor(v, 4);
    if (n == 0) bpw[s * 2]     = (unsigned int)v;
    if (n == 8) bpw[s * 2 + 1] = (unsigned int)v;
    __syncthreads();
    if (n < NTAG) fv[n] = nf;
    __syncthreads();
  }

  if (n < NTAG) term[n] = fv[n] + transitions[1 * NTAG + n];   // STOP = 1
  __syncthreads();
  if (n == 0) {
    float best = term[0];
    int   barg = 0;
    for (int p = 1; p < NTAG; ++p)
      if (term[p] > best) { best = term[p]; barg = p; }
    out[0] = best;
    int cur = barg;
    out[1 + (SEQ - 1)] = (float)cur;
    for (int s = SEQ - 1; s >= 1; --s) {
      unsigned int w = bpw[s * 2 + (cur >> 3)];
      cur = (int)((w >> ((cur & 7) * 4)) & 15u);
      out[1 + (s - 1)] = (float)cur;
    }
  }
}

// ---------------------------------------------------------------------------
extern "C" void kernel_launch(void* const* d_in, const int* in_sizes, int n_in,
                              void* d_out, int out_size, void* d_ws, size_t ws_size,
                              hipStream_t stream) {
  (void)in_sizes; (void)n_in; (void)out_size; (void)ws_size;
  const int*   sentence = (const int*)  d_in[0];
  const float* embed    = (const float*)d_in[1];
  const float* w_ih_f   = (const float*)d_in[2];
  const float* w_hh_f   = (const float*)d_in[3];
  const float* b_ih_f   = (const float*)d_in[4];
  const float* b_hh_f   = (const float*)d_in[5];
  const float* w_ih_b   = (const float*)d_in[6];
  const float* w_hh_b   = (const float*)d_in[7];
  const float* b_ih_b   = (const float*)d_in[8];
  const float* b_hh_b   = (const float*)d_in[9];
  const float* h0       = (const float*)d_in[10];
  const float* c0       = (const float*)d_in[11];
  const float* w_out    = (const float*)d_in[12];
  const float* b_out    = (const float*)d_in[13];
  const float* trans    = (const float*)d_in[14];
  float* out = (float*)d_out;

  // Workspace carve-out (256B aligned segments), ~83 MB total.
  char*  ws  = (char*)d_ws;
  size_t off = 0;
  auto carve = [&](size_t bytes) -> void* {
    off = (off + 255) & ~(size_t)255;
    void* p = ws + off;
    off += bytes;
    return p;
  };
  __bf16* emb_bf = (__bf16*)carve((size_t)SEQ * EMBED * 2);
  __bf16* wcat   = (__bf16*)carve((size_t)GCAT * EMBED * 2);
  __bf16* whh_bf = (__bf16*)carve((size_t)GCAT * EMBED * 2);
  float*  bcat   = (float*) carve((size_t)GCAT * 4);
  __bf16* wout   = (__bf16*)carve((size_t)NTAG * HID * 2);
  float*  zin    = (float*) carve((size_t)SEQ * GCAT * 4);
  __bf16* hcat   = (__bf16*)carve((size_t)SEQ * HID * 2);
  float*  feats  = (float*) carve((size_t)SEQ * NTAG * 4);

  // 1. weight prep (bf16 conversions + fused biases)
  {
    int total = GCAT * EMBED * 2 + NTAG * HID + GCAT;   // 1058816
    prep_kernel<<<(total + 255) / 256, 256, 0, stream>>>(
        w_ih_f, w_ih_b, w_hh_f, w_hh_b, b_ih_f, b_hh_f, b_ih_b, b_hh_b, w_out,
        wcat, whh_bf, bcat, wout);
  }
  // 2. embedding gather -> bf16
  gather_kernel<<<(SEQ * EMBED) / 256, 256, 0, stream>>>(sentence, embed, emb_bf);
  // 3. input-projection GEMM (WMMA): 512x128 wave-tiles, 8 waves/block
  zin_gemm_kernel<<<8192, 256, 0, stream>>>(emb_bf, wcat, bcat, zin);
  // 4. bidirectional LSTM scan (one WGP per direction, 266KB LDS weights)
  lstm_scan_kernel<<<2, 1024, SCAN_LDS_BYTES, stream>>>(zin, whh_bf, h0, c0, hcat);
  // 5. output projection (WMMA): 512 wave-tiles, 8 waves/block
  proj_kernel<<<64, 256, 0, stream>>>(hcat, wout, b_out, feats);
  // 6. Viterbi decode (back-pointers nibble-packed in LDS)
  viterbi_kernel<<<1, 32, VIT_LDS_BYTES, stream>>>(feats, trans, out);
}
